// MultiHeadSelfAttention_52123723105032
// MI455X (gfx1250) — matrix operand order repaired and checked by reading, not compiled, not run on hardware
//
#include <hip/hip_runtime.h>
#include <hip/hip_bf16.h>
#include <stdint.h>

typedef __attribute__((ext_vector_type(16))) __bf16 v16bf;
typedef __attribute__((ext_vector_type(8)))  float  v8f;
typedef __attribute__((ext_vector_type(4)))  unsigned int u32x4_t;
typedef __attribute__((ext_vector_type(8)))  int i32x8_t;
typedef __attribute__((ext_vector_type(4)))  int i32x4_t;

union FragA { uint4 u4[2]; v16bf v; };

__device__ inline unsigned short f32_bf16(float f) {
    union { float f; unsigned u; } x; x.f = f;
    unsigned r = x.u + 0x7FFFu + ((x.u >> 16) & 1u);
    return (unsigned short)(r >> 16);
}

__device__ inline v8f wmma_bf16(const v16bf& a, const v16bf& b, const v8f& c) {
    return __builtin_amdgcn_wmma_f32_16x16x32_bf16(false, a, false, b, (short)0, c, false, false);
}

#if __has_builtin(__builtin_amdgcn_tensor_load_to_lds)
#define USE_TDM 1
#else
#define USE_TDM 0
#endif

#if USE_TDM
// 2-D tile DMA: 64x64 bf16 tile from a row-major [tens_h][stride] bf16 tensor.
// D# per CDNA5 ISA ch.8: group0 = {count=1 | lds_addr | global_addr | type=2},
// group1 = {data_size=1(2B), tensor_dim0/1, tile_dim0/1=64, dim0 stride}.
__device__ inline void tdm_load_tile(unsigned lds_off, const unsigned short* g,
                                     unsigned tens_w, unsigned tens_h, unsigned stride) {
    unsigned long long ga = (unsigned long long)(uintptr_t)g;
    u32x4_t g0;
    g0[0] = 1u;                                            // count=1 (valid), user mode
    g0[1] = lds_off;                                       // LDS byte address
    g0[2] = (unsigned)ga;                                  // global_addr[31:0]
    g0[3] = ((unsigned)(ga >> 32) & 0x01FFFFFFu) | 0x80000000u; // addr[56:32] | type=2
    i32x8_t g1;
    g1[0] = 1 << 16;                                       // wg_mask=0, data_size=1 (2B)
    g1[1] = (int)((tens_w & 0xFFFFu) << 16);               // tensor_dim0 lo
    g1[2] = (int)((tens_w >> 16) | ((tens_h & 0xFFFFu) << 16)); // dim0 hi | dim1 lo
    g1[3] = (int)((tens_h >> 16) | (64u << 16));           // dim1 hi | tile_dim0=64
    g1[4] = 64;                                            // tile_dim1=64, tile_dim2=0
    g1[5] = (int)stride;                                   // tensor_dim0_stride lo
    g1[6] = 0;                                             // stride hi | dim1_stride lo
    g1[7] = 0;
    i32x4_t z4 = (i32x4_t)0;
#if defined(__clang_major__) && (__clang_major__ >= 23)
    i32x8_t z8 = (i32x8_t)0;
    __builtin_amdgcn_tensor_load_to_lds(g0, g1, z4, z4, z8, 0);
#else
    __builtin_amdgcn_tensor_load_to_lds(g0, g1, z4, z4, 0);
#endif
}
#endif

// ---------------------------------------------------------------------------
// One-time f32 -> bf16 conversion kernels
// ---------------------------------------------------------------------------
__global__ __launch_bounds__(256) void cvt_flat(const float* __restrict__ s,
                                                unsigned short* __restrict__ d, int n4) {
    int i = blockIdx.x * 256 + threadIdx.x;
    if (i < n4) {
        const float4 f = ((const float4*)s)[i];
        uint2 o;
        o.x = (unsigned)f32_bf16(f.x) | ((unsigned)f32_bf16(f.y) << 16);
        o.y = (unsigned)f32_bf16(f.z) | ((unsigned)f32_bf16(f.w) << 16);
        ((uint2*)d)[i] = o;
    }
}

// src [rows][cols] f32 -> dst [cols][rows] bf16 (tiled transpose via LDS)
__global__ __launch_bounds__(256) void cvt_transpose(const float* __restrict__ src,
                                                     unsigned short* __restrict__ dst,
                                                     int rows, int cols) {
    __shared__ unsigned short tile[32][33];
    const int tx = threadIdx.x & 31, ty = threadIdx.x >> 5;
    const int r0 = blockIdx.y * 32, c0 = blockIdx.x * 32;
    for (int i = ty; i < 32; i += 8)
        tile[i][tx] = f32_bf16(src[(size_t)(r0 + i) * cols + c0 + tx]);
    __syncthreads();
    for (int i = ty; i < 32; i += 8)
        dst[(size_t)(c0 + i) * rows + r0 + tx] = tile[tx][i];
}

// ---------------------------------------------------------------------------
// Shared GEMM inner step: 64x64 block tile, wave (wm,wn) does 32x32, K-chunk 64
// ---------------------------------------------------------------------------
__device__ inline void gemm_step(const unsigned short (*As)[64], const unsigned short (*Bs)[64],
                                 int wm, int wn, int half, int l15, v8f acc[2][2]) {
    for (int ks = 0; ks < 64; ks += 32) {
        FragA a[2], b[2];
        for (int mi = 0; mi < 2; ++mi) {
            const unsigned short* p = &As[wm * 32 + mi * 16 + l15][ks];
            a[mi].u4[0] = *(const uint4*)(p + half * 8);
            a[mi].u4[1] = *(const uint4*)(p + 16 + half * 8);
        }
        for (int ni = 0; ni < 2; ++ni) {
            const unsigned short* p = &Bs[wn * 32 + ni * 16 + l15][ks];
            b[ni].u4[0] = *(const uint4*)(p + half * 8);
            b[ni].u4[1] = *(const uint4*)(p + 16 + half * 8);
        }
        for (int mi = 0; mi < 2; ++mi)
            for (int ni = 0; ni < 2; ++ni)
                acc[mi][ni] = wmma_bf16(a[mi].v, b[ni].v, acc[mi][ni]);
    }
}

// ---------------------------------------------------------------------------
// Kernel 1: qkv = xb @ WqkvT^T + bqkv -> bf16 Q(scaled)/K/V in [B,H,L,64]
// ---------------------------------------------------------------------------
__global__ __launch_bounds__(128) void qkv_gemm(const unsigned short* __restrict__ xb,
                                                const unsigned short* __restrict__ Wt,
                                                const float* __restrict__ bias,
                                                unsigned short* __restrict__ qws,
                                                unsigned short* __restrict__ kws,
                                                unsigned short* __restrict__ vws) {
    __shared__ unsigned short As[2][64][64];
    __shared__ unsigned short Bs[2][64][64];
    const int t = threadIdx.x, lane = t & 31, wave = t >> 5;
    const int wm = wave >> 1, wn = wave & 1;
    const int half = lane >> 4, l15 = lane & 15;
    const int m0 = blockIdx.y * 64, n0 = blockIdx.x * 64;
    const int NC = 1024 / 64;
    v8f acc[2][2] = {};

#if USE_TDM
    if (wave == 0) {
        tdm_load_tile((unsigned)(uintptr_t)&As[0][0][0], xb + (size_t)m0 * 1024, 1024, 8192, 1024);
        tdm_load_tile((unsigned)(uintptr_t)&Bs[0][0][0], Wt + (size_t)n0 * 1024, 1024, 3072, 1024);
    }
#endif
    for (int kc = 0; kc < NC; ++kc) {
        const int buf = kc & 1;
        const int k0 = kc * 64;
#if USE_TDM
        if (wave == 0) {
            if (kc + 1 < NC) {
                tdm_load_tile((unsigned)(uintptr_t)&As[buf ^ 1][0][0],
                              xb + (size_t)m0 * 1024 + k0 + 64, 1024, 8192, 1024);
                tdm_load_tile((unsigned)(uintptr_t)&Bs[buf ^ 1][0][0],
                              Wt + (size_t)n0 * 1024 + k0 + 64, 1024, 3072, 1024);
                __builtin_amdgcn_s_wait_tensorcnt(2);   // current pair done, next in flight
            } else {
                __builtin_amdgcn_s_wait_tensorcnt(0);
            }
        }
#else
        {
            unsigned* da = (unsigned*)&As[buf][0][0];
            unsigned* db = (unsigned*)&Bs[buf][0][0];
            for (int i = 0; i < 16; ++i) {
                int e = t + i * 128, row = e >> 5, c = e & 31;
                da[e] = ((const unsigned*)(xb + (size_t)(m0 + row) * 1024 + k0))[c];
                db[e] = ((const unsigned*)(Wt + (size_t)(n0 + row) * 1024 + k0))[c];
            }
        }
#endif
        __syncthreads();
        gemm_step(As[buf], Bs[buf], wm, wn, half, l15, acc);
        __syncthreads();
    }
    // epilogue: bias, Q scale 1/sqrt(64), scatter to [B,H,L,64] bf16
    for (int mi = 0; mi < 2; ++mi)
        for (int ni = 0; ni < 2; ++ni) {
            int ng = n0 + wn * 32 + ni * 16 + l15;
            float bv = bias[ng];
            int which = ng >> 10, d = ng & 1023, h = d >> 6, hd = d & 63;
            unsigned short* dst = (which == 0) ? qws : ((which == 1) ? kws : vws);
            float scl = (which == 0) ? 0.125f : 1.0f;
            for (int r = 0; r < 8; ++r) {
                int m = m0 + wm * 32 + mi * 16 + r + 8 * half;
                int b_ = m >> 11, l_ = m & 2047;
                dst[((size_t)(b_ * 16 + h) * 2048 + l_) * 64 + hd] =
                    f32_bf16((acc[mi][ni][r] + bv) * scl);
            }
        }
}

// ---------------------------------------------------------------------------
// Kernel 2: causal flash attention. 1 block = (b,h) x 64 q rows, 4 waves.
// ---------------------------------------------------------------------------
__global__ __launch_bounds__(128) void attn(const unsigned short* __restrict__ Q,
                                            const unsigned short* __restrict__ K,
                                            const unsigned short* __restrict__ V,
                                            unsigned short* __restrict__ Y) {
    __shared__ unsigned short Qs[64][64];
    __shared__ unsigned short Ks[64][64];
    __shared__ unsigned short Vt[64][64];      // transposed: [hd][key]
    __shared__ unsigned short Ps[4][16][64];   // per-wave P tile
    const int t = threadIdx.x, lane = t & 31, wave = t >> 5;
    const int half = lane >> 4, l15 = lane & 15;
    const int qb0 = blockIdx.x * 64;
    const int bh = blockIdx.y, b = bh >> 4, h = bh & 15;
    const size_t base = (size_t)bh * 2048 * 64;

    {
        const unsigned int* Qg = (const unsigned int*)(Q + base + (size_t)qb0 * 64);
        unsigned int* Qs32 = (unsigned int*)&Qs[0][0];
        for (int i = 0; i < 16; ++i) Qs32[t + i * 128] = Qg[t + i * 128];
    }
    __syncthreads();
    FragA aq[2];
    {
        const unsigned short* p = &Qs[wave * 16 + l15][0];
        aq[0].u4[0] = *(const uint4*)(p + half * 8);
        aq[0].u4[1] = *(const uint4*)(p + 16 + half * 8);
        aq[1].u4[0] = *(const uint4*)(p + 32 + half * 8);
        aq[1].u4[1] = *(const uint4*)(p + 48 + half * 8);
    }
    v8f o[4] = {};
    float mrow[8], lrow[8];
    for (int r = 0; r < 8; ++r) { mrow[r] = -3.0e38f; lrow[r] = 0.f; }

    for (int kb = 0; kb < qb0 + 64; kb += 64) {
        {
            const unsigned int* Kg = (const unsigned int*)(K + base + (size_t)kb * 64);
            unsigned int* Ks32 = (unsigned int*)&Ks[0][0];
            for (int i = 0; i < 16; ++i) Ks32[t + i * 128] = Kg[t + i * 128];
            const unsigned int* Vg = (const unsigned int*)(V + base + (size_t)kb * 64);
            for (int i = 0; i < 16; ++i) {
                int e = t + i * 128;
                int key = e >> 5, hp = e & 31;
                unsigned int w = Vg[e];
                Vt[2 * hp][key]     = (unsigned short)(w & 0xFFFFu);
                Vt[2 * hp + 1][key] = (unsigned short)(w >> 16);
            }
        }
        __syncthreads();

        v8f s[4];
        for (int kg = 0; kg < 4; ++kg) {
            FragA bk0, bk1;
            const unsigned short* p = &Ks[kg * 16 + l15][0];
            bk0.u4[0] = *(const uint4*)(p + half * 8);
            bk0.u4[1] = *(const uint4*)(p + 16 + half * 8);
            bk1.u4[0] = *(const uint4*)(p + 32 + half * 8);
            bk1.u4[1] = *(const uint4*)(p + 48 + half * 8);
            v8f z = {};
            s[kg] = wmma_bf16(aq[0].v, bk0.v, z);
            s[kg] = wmma_bf16(aq[1].v, bk1.v, s[kg]);
        }
        for (int r = 0; r < 8; ++r) {
            int q = qb0 + wave * 16 + r + 8 * half;
            float mx = -3.0e38f;
            for (int kg = 0; kg < 4; ++kg) {
                float val = s[kg][r];
                if (kb + kg * 16 + l15 > q) { val = -3.0e38f; s[kg][r] = val; }
                mx = fmaxf(mx, val);
            }
            for (int mk = 1; mk < 16; mk <<= 1) mx = fmaxf(mx, __shfl_xor(mx, mk, 32));
            float mnew = fmaxf(mrow[r], mx);
            float scalef = __expf(mrow[r] - mnew);
            o[0][r] *= scalef; o[1][r] *= scalef; o[2][r] *= scalef; o[3][r] *= scalef;
            float rs = 0.f;
            for (int kg = 0; kg < 4; ++kg) {
                float pv = __expf(s[kg][r] - mnew);
                s[kg][r] = pv;
                rs += pv;
            }
            for (int mk = 1; mk < 16; mk <<= 1) rs += __shfl_xor(rs, mk, 32);
            lrow[r] = lrow[r] * scalef + rs;
            mrow[r] = mnew;
        }
        for (int kg = 0; kg < 4; ++kg)
            for (int r = 0; r < 8; ++r)
                Ps[wave][r + 8 * half][kg * 16 + l15] = f32_bf16(s[kg][r]);
        FragA ap[2];
        {
            const unsigned short* p = &Ps[wave][l15][0];
            ap[0].u4[0] = *(const uint4*)(p + half * 8);
            ap[0].u4[1] = *(const uint4*)(p + 16 + half * 8);
            ap[1].u4[0] = *(const uint4*)(p + 32 + half * 8);
            ap[1].u4[1] = *(const uint4*)(p + 48 + half * 8);
        }
        for (int nt = 0; nt < 4; ++nt) {
            const unsigned short* p = &Vt[nt * 16 + l15][0];
            for (int cc = 0; cc < 2; ++cc) {
                FragA bv;
                bv.u4[0] = *(const uint4*)(p + cc * 32 + half * 8);
                bv.u4[1] = *(const uint4*)(p + cc * 32 + 16 + half * 8);
                o[nt] = wmma_bf16(ap[cc].v, bv.v, o[nt]);
            }
        }
        __syncthreads();
    }
    for (int r = 0; r < 8; ++r) {
        float inv = 1.0f / lrow[r];
        int q = qb0 + wave * 16 + r + 8 * half;
        size_t rowoff = ((size_t)(b * 2048 + q)) * 1024 + h * 64;
        for (int nt = 0; nt < 4; ++nt)
            Y[rowoff + nt * 16 + l15] = f32_bf16(o[nt][r] * inv);
    }
}

// ---------------------------------------------------------------------------
// Kernel 3: out = Y @ WprojT^T + bproj (f32 output)
// ---------------------------------------------------------------------------
__global__ __launch_bounds__(128) void proj_gemm(const unsigned short* __restrict__ Yb,
                                                 const unsigned short* __restrict__ Wt,
                                                 const float* __restrict__ bias,
                                                 float* __restrict__ out) {
    __shared__ unsigned short As[2][64][64];
    __shared__ unsigned short Bs[2][64][64];
    const int t = threadIdx.x, lane = t & 31, wave = t >> 5;
    const int wm = wave >> 1, wn = wave & 1;
    const int half = lane >> 4, l15 = lane & 15;
    const int m0 = blockIdx.y * 64, n0 = blockIdx.x * 64;
    const int NC = 1024 / 64;
    v8f acc[2][2] = {};

#if USE_TDM
    if (wave == 0) {
        tdm_load_tile((unsigned)(uintptr_t)&As[0][0][0], Yb + (size_t)m0 * 1024, 1024, 8192, 1024);
        tdm_load_tile((unsigned)(uintptr_t)&Bs[0][0][0], Wt + (size_t)n0 * 1024, 1024, 1024, 1024);
    }
#endif
    for (int kc = 0; kc < NC; ++kc) {
        const int buf = kc & 1;
        const int k0 = kc * 64;
#if USE_TDM
        if (wave == 0) {
            if (kc + 1 < NC) {
                tdm_load_tile((unsigned)(uintptr_t)&As[buf ^ 1][0][0],
                              Yb + (size_t)m0 * 1024 + k0 + 64, 1024, 8192, 1024);
                tdm_load_tile((unsigned)(uintptr_t)&Bs[buf ^ 1][0][0],
                              Wt + (size_t)n0 * 1024 + k0 + 64, 1024, 1024, 1024);
                __builtin_amdgcn_s_wait_tensorcnt(2);
            } else {
                __builtin_amdgcn_s_wait_tensorcnt(0);
            }
        }
#else
        {
            unsigned* da = (unsigned*)&As[buf][0][0];
            unsigned* db = (unsigned*)&Bs[buf][0][0];
            for (int i = 0; i < 16; ++i) {
                int e = t + i * 128, row = e >> 5, c = e & 31;
                da[e] = ((const unsigned*)(Yb + (size_t)(m0 + row) * 1024 + k0))[c];
                db[e] = ((const unsigned*)(Wt + (size_t)(n0 + row) * 1024 + k0))[c];
            }
        }
#endif
        __syncthreads();
        gemm_step(As[buf], Bs[buf], wm, wn, half, l15, acc);
        __syncthreads();
    }
    for (int mi = 0; mi < 2; ++mi)
        for (int ni = 0; ni < 2; ++ni) {
            int ng = n0 + wn * 32 + ni * 16 + l15;
            float bv = bias[ng];
            for (int r = 0; r < 8; ++r) {
                int m = m0 + wm * 32 + mi * 16 + r + 8 * half;
                out[(size_t)m * 1024 + ng] = acc[mi][ni][r] + bv;
            }
        }
}

extern "C" void kernel_launch(void* const* d_in, const int* in_sizes, int n_in,
                              void* d_out, int out_size, void* d_ws, size_t ws_size,
                              hipStream_t stream) {
    const float* x     = (const float*)d_in[0];
    const float* Wqkv  = (const float*)d_in[1];
    const float* bqkv  = (const float*)d_in[2];
    const float* Wproj = (const float*)d_in[3];
    const float* bproj = (const float*)d_in[4];
    float* out = (float*)d_out;

    const size_t NE_X  = (size_t)8192 * 1024;   // 8,388,608
    const size_t NE_WQ = (size_t)3072 * 1024;
    const size_t NE_WP = (size_t)1024 * 1024;
    unsigned short* xb  = (unsigned short*)d_ws;
    unsigned short* wqt = xb + NE_X;
    unsigned short* wpt = wqt + NE_WQ;
    unsigned short* qws = wpt + NE_WP;
    unsigned short* kws = qws + NE_X;
    unsigned short* vws = kws + NE_X;
    unsigned short* yws = vws + NE_X;

    cvt_flat<<<8192, 256, 0, stream>>>(x, xb, (int)(NE_X / 4));
    cvt_transpose<<<dim3(96, 32), 256, 0, stream>>>(Wqkv, wqt, 1024, 3072);
    cvt_transpose<<<dim3(32, 32), 256, 0, stream>>>(Wproj, wpt, 1024, 1024);
    qkv_gemm<<<dim3(48, 128), 128, 0, stream>>>(xb, wqt, bqkv, qws, kws, vws);
    attn<<<dim3(32, 64), 128, 0, stream>>>(qws, kws, vws, yws);
    proj_gemm<<<dim3(16, 128), 128, 0, stream>>>(yws, wpt, bproj, out);
}
